// TemporalContrastiveLoss_13872744366784
// MI455X (gfx1250) — matrix operand order repaired and checked by reading, not compiled, not run on hardware
//
#include <hip/hip_runtime.h>

// ---------------------------------------------------------------------------
// TemporalContrastiveLoss for MI455X (gfx1250, wave32, WMMA).
//
// Math notes (derived from the reference):
//   positive_mask = (1-done_mask)*(1-temporal) + eye == 1 - temporal + eye
//   (done_mask is annihilated by its own block-diagonal factor), so `done`
//   is mathematically unused, neg[r,c] = same-trajectory-block && c!=r, and
//   sum(neg,axis=1) == 2*T-1 == 31 for every row.
//   Normalized rows => logits <= 1/TEMP = 10, so softmax can use a FIXED
//   shift of 10 (no row-max pass): LSE[r] = 10 + log(sum_c exp(logit-10)).
//   loss = -mean_r[ (S[r] - 31*LSE[r]) / 31 ],  S[r] = sum_{neg} logit.
//
// GEMM: zn@zn^T with bf16 "hi+lo" error-compensated split:
//   dot(a,b) ~= ah.bh + ah.bl + al.bh   (3x v_wmma_f32_16x16x32_bf16)
// ---------------------------------------------------------------------------

#define MM   4096   // 2*N*T
#define DD   512
#define NTT  2048   // N*T
#define TSPAN 16
#define INV_TEMP 10.0f
#define SHIFT    10.0f
#define NEGCNT   31.0f

typedef __attribute__((ext_vector_type(16))) __bf16 v16bf;
typedef __attribute__((ext_vector_type(8)))  float  v8f;

union FragBF { v16bf v; unsigned int u[8]; };

__device__ __forceinline__ unsigned short f32_to_bf16_rne(float f) {
    unsigned int u = __float_as_uint(f);
    u += 0x7FFFu + ((u >> 16) & 1u);   // round-to-nearest-even
    return (unsigned short)(u >> 16);
}
__device__ __forceinline__ float bf16_to_f32(unsigned short h) {
    return __uint_as_float(((unsigned int)h) << 16);
}

// ---------------------------------------------------------------------------
// K1: row-normalize z (f32), split into bf16 hi/lo planes; zero E/S accums.
// One wave32 per row; 8 rows per 256-thread block.
// ---------------------------------------------------------------------------
__global__ void tcl_normalize_split(const float* __restrict__ z,
                                    unsigned short* __restrict__ zh,
                                    unsigned short* __restrict__ zl,
                                    float* __restrict__ E,
                                    float* __restrict__ S) {
    const int wave = threadIdx.x >> 5;
    const int lane = threadIdx.x & 31;
    const int row  = blockIdx.x * 8 + wave;

    const float* zr = z + (size_t)row * DD;
    float acc = 0.0f;
#pragma unroll
    for (int i = lane; i < DD; i += 32) {
        float v = zr[i];
        acc += v * v;
    }
#pragma unroll
    for (int m = 16; m >= 1; m >>= 1)
        acc += __shfl_xor(acc, m, 32);

    float norm = fmaxf(sqrtf(acc), 1e-8f);
    float rinv = 1.0f / norm;

    unsigned short* hr = zh + (size_t)row * DD;
    unsigned short* lr = zl + (size_t)row * DD;
#pragma unroll
    for (int i = lane; i < DD; i += 32) {
        float zn = zr[i] * rinv;
        unsigned short h = f32_to_bf16_rne(zn);
        float lo = zn - bf16_to_f32(h);
        hr[i] = h;
        lr[i] = f32_to_bf16_rne(lo);
    }
    if (lane == 0) { E[row] = 0.0f; S[row] = 0.0f; }
}

// ---------------------------------------------------------------------------
// K2: fused Gram-matrix GEMM + softmax partial reductions.
// Grid (64,64): each workgroup computes a 64x64 logits tile.
// 4 waves/WG; wave w owns rows [bm*64+w*16, +16) x all 64 cols
// => 4 f32 16x16 accumulators, A-fragments reused across the 4 column tiles.
//
// Fragment lane layouts per CDNA5 ISA 7.12.2 (wave32):
//   A 16x32 bf16: lane L: M = L%16; dword j: K = 2j + (L>=16 ? 8 : 0) (j<4)
//                                           K = 2j + 8 + (L>=16 ? 8 : 0) (j>=4)
//   B 32x16 bf16: lane L: N = L%16; dword v: same K mapping as A (v<4: 2v, v>=4: 2v+8; +8 if L>=16)
//   C/D 16x16 f32: lane L, vgpr v: M = v + (L>=16 ? 8 : 0), N = L%16
// ---------------------------------------------------------------------------
__global__ void __launch_bounds__(128)
tcl_gram_softmax(const unsigned short* __restrict__ zh,
                 const unsigned short* __restrict__ zl,
                 float* __restrict__ E,
                 float* __restrict__ S) {
    const int wave = threadIdx.x >> 5;
    const int lane = threadIdx.x & 31;
    const int lh   = lane >> 4;   // lane half (0/1)
    const int lm   = lane & 15;   // position within half

    const int row_base = blockIdx.x * 64 + wave * 16;
    const int col_base = blockIdx.y * 64;

    v8f acc0 = (v8f)0.0f, acc1 = (v8f)0.0f, acc2 = (v8f)0.0f, acc3 = (v8f)0.0f;

    const size_t arow = (size_t)(row_base + lm) * DD;

    for (int k0 = 0; k0 < DD; k0 += 32) {
        FragBF ah, al;
#pragma unroll
        for (int j = 0; j < 8; ++j) {
            const int kk = ((j < 4) ? (2 * j) : (2 * j + 8)) + lh * 8;
            ah.u[j] = *(const unsigned int*)(zh + arow + k0 + kk);
            al.u[j] = *(const unsigned int*)(zl + arow + k0 + kk);
        }
#pragma unroll
        for (int c = 0; c < 4; ++c) {
            const size_t brow = (size_t)(col_base + c * 16 + lm) * DD;
            FragBF bh, bl;
#pragma unroll
            for (int v = 0; v < 8; ++v) {
                const int kk = ((v < 4) ? (2 * v) : (2 * v + 8)) + lh * 8;
                bh.u[v] = *(const unsigned int*)(zh + brow + k0 + kk);
                bl.u[v] = *(const unsigned int*)(zl + brow + k0 + kk);
            }
            v8f a = (c == 0) ? acc0 : (c == 1) ? acc1 : (c == 2) ? acc2 : acc3;
            a = __builtin_amdgcn_wmma_f32_16x16x32_bf16(false, ah.v, false, bh.v,
                                                        (short)0, a, false, false);
            a = __builtin_amdgcn_wmma_f32_16x16x32_bf16(false, ah.v, false, bl.v,
                                                        (short)0, a, false, false);
            a = __builtin_amdgcn_wmma_f32_16x16x32_bf16(false, al.v, false, bh.v,
                                                        (short)0, a, false, false);
            if (c == 0) acc0 = a; else if (c == 1) acc1 = a;
            else if (c == 2) acc2 = a; else acc3 = a;
        }
    }

    // Epilogue: exp-sum (fixed shift 10) + in-block logit sum, per output row.
#pragma unroll
    for (int v = 0; v < 8; ++v) {
        const int r  = row_base + v + lh * 8;
        const int tr = (r & (NTT - 1)) >> 4;          // trajectory of row r
        float esum = 0.0f, ssum = 0.0f;
#pragma unroll
        for (int c = 0; c < 4; ++c) {
            const float raw = (c == 0) ? acc0[v] : (c == 1) ? acc1[v]
                            : (c == 2) ? acc2[v] : acc3[v];
            const int col = col_base + c * 16 + lm;
            if (col != r) {
                const float logit = raw * INV_TEMP;
                esum += __expf(logit - SHIFT);
                if (((col & (NTT - 1)) >> 4) == tr)   // same trajectory block
                    ssum += logit;
            }
        }
        // reduce across the 16 lanes of this half (masks 8..1 stay in-half)
#pragma unroll
        for (int m = 8; m >= 1; m >>= 1) {
            esum += __shfl_xor(esum, m, 32);
            ssum += __shfl_xor(ssum, m, 32);
        }
        if (lm == 0) {
            atomicAdd(&E[r], esum);   // global_atomic_add_f32
            atomicAdd(&S[r], ssum);
        }
    }
}

// ---------------------------------------------------------------------------
// K3: loss = -(1/M) * sum_r (S[r] - 31*(SHIFT + log E[r])) / 31
// ---------------------------------------------------------------------------
__global__ void tcl_finalize(const float* __restrict__ E,
                             const float* __restrict__ S,
                             float* __restrict__ out) {
    __shared__ float sdata[256];
    float local = 0.0f;
    for (int r = threadIdx.x; r < MM; r += 256) {
        float lse = SHIFT + __logf(E[r]);
        local += (S[r] - NEGCNT * lse) * (1.0f / NEGCNT);
    }
    sdata[threadIdx.x] = local;
    __syncthreads();
    for (int s = 128; s > 0; s >>= 1) {
        if (threadIdx.x < s) sdata[threadIdx.x] += sdata[threadIdx.x + s];
        __syncthreads();
    }
    if (threadIdx.x == 0) out[0] = -sdata[0] / (float)MM;
}

extern "C" void kernel_launch(void* const* d_in, const int* in_sizes, int n_in,
                              void* d_out, int out_size, void* d_ws, size_t ws_size,
                              hipStream_t stream) {
    (void)in_sizes; (void)n_in; (void)out_size; (void)ws_size;

    const float* z = (const float*)d_in[0];
    // d_in[1] (`done`) is provably unused: its mask term is annihilated by
    // the block-diagonal temporal mask in the reference's positive_mask.

    // workspace: bf16 hi plane (4MB) | bf16 lo plane (4MB) | E[M] | S[M]
    unsigned short* zh = (unsigned short*)d_ws;
    unsigned short* zl = zh + (size_t)MM * DD;
    float*          E  = (float*)(zl + (size_t)MM * DD);
    float*          S  = E + MM;
    float*          out = (float*)d_out;

    tcl_normalize_split<<<MM / 8, 256, 0, stream>>>(z, zh, zl, E, S);
    tcl_gram_softmax<<<dim3(MM / 64, MM / 64), 128, 0, stream>>>(zh, zl, E, S);
    tcl_finalize<<<1, 256, 0, stream>>>(E, S, out);
}
